// SpConAtt_72859825209510
// MI455X (gfx1250) — matrix operand order repaired and checked by reading, not compiled, not run on hardware
//
#include <hip/hip_runtime.h>
#include <hip/hip_bf16.h>

// ---------------------------------------------------------------------------
// SpConAtt on MI455X (gfx1250): bf16 WMMA pipeline + online softmax + TDM
// gather for the windowed V rows.  B=4, CIN=256, H=W=64, C=64, K=7.
// ---------------------------------------------------------------------------

typedef __attribute__((ext_vector_type(16))) __bf16 v16bf;
typedef __attribute__((ext_vector_type(8)))  __bf16 v8bf;
typedef __attribute__((ext_vector_type(8)))  float  v8f;
typedef __attribute__((ext_vector_type(4)))  unsigned int v4u;
typedef __attribute__((ext_vector_type(8)))  int v8i;
typedef __attribute__((ext_vector_type(4)))  int v4i;

#define NB    4
#define CIN_  256
#define HWQ   4096
#define CMID  64

#if defined(__has_builtin)
#if __has_builtin(__builtin_amdgcn_tensor_load_to_lds) && \
    __has_builtin(__builtin_amdgcn_s_wait_tensorcnt)
#define HAS_TDM 1
#endif
#endif
#ifndef HAS_TDM
#define HAS_TDM 0
#endif

static __device__ __forceinline__ v8f wmma_bf16(v16bf a, v16bf b, v8f c) {
    return __builtin_amdgcn_wmma_f32_16x16x32_bf16(false, a, false, b,
                                                   (short)0, c, false, false);
}

static __device__ __forceinline__ v16bf cat8(v8bf lo, v8bf hi) {
    v16bf r;
#pragma unroll
    for (int i = 0; i < 8; ++i) { r[i] = lo[i]; r[i + 8] = hi[i]; }
    return r;
}

#if HAS_TDM
// TDM gather-mode load: 16 rows of `width=64` bf16 elements, rows selected by
// 16-bit indices, packed consecutively (128B/row) at LDS byte offset lds_byte.
// D# layout per CDNA5 ISA.
static __device__ __forceinline__ void tdm_gather16(
    unsigned lds_byte, const __bf16* base, const unsigned short* idx)
{
    unsigned long long ga = (unsigned long long)(uintptr_t)base;
    v4u g0;
    g0[0] = 1u | (1u << 31);                 // count=1, gather_mode=1, idx16
    g0[1] = lds_byte;                        // lds_addr (bytes)
    g0[2] = (unsigned)(ga & 0xffffffffu);    // global_addr[31:0]
    g0[3] = (unsigned)((ga >> 32) & 0x01ffffffu) | (2u << 30); // addr[56:32],type=2
    v8i g1;
    g1[0] = (1 << 16);        // data_size = 2 bytes
    g1[1] = (64 << 16);       // tensor_dim0 = 64 (elements)
    g1[2] = (4096 << 16);     // tensor_dim1 = 4096 (rows)
    g1[3] = (64 << 16);       // tile_dim0 = 64 (row width)
    g1[4] = 16;               // tile_dim1 = #valid indices
    g1[5] = 64;               // tensor_dim0_stride = 64
    g1[6] = 0;
    g1[7] = 0;
    v4i g2, g3;
#pragma unroll
    for (int e = 0; e < 4; ++e) {
        g2[e] = (int)((unsigned)idx[2 * e]     | ((unsigned)idx[2 * e + 1] << 16));
        g3[e] = (int)((unsigned)idx[2 * e + 8] | ((unsigned)idx[2 * e + 9] << 16));
    }
#if __clang_major__ >= 23
    v8i gz = {};
    __builtin_amdgcn_tensor_load_to_lds(g0, g1, g2, g3, gz, 0);
#else
    __builtin_amdgcn_tensor_load_to_lds(g0, g1, g2, g3, 0);
#endif
}
#endif

// ---------------------------------------------------------------------------
// Kernel 0: repack Wout(o,c,ki,kj) f32 -> Bt[w49][o][c] bf16
// ---------------------------------------------------------------------------
__global__ void spc_prep(const float* __restrict__ Wout, __bf16* __restrict__ Bt) {
    int idx = blockIdx.x * blockDim.x + threadIdx.x;
    if (idx >= 49 * 64 * 64) return;
    int c  = idx & 63;
    int o  = (idx >> 6) & 63;
    int w  = idx >> 12;
    int di = w / 7, dj = w % 7;
    Bt[idx] = (__bf16)Wout[((o * 64 + c) * 7 + di) * 7 + dj];
}

// ---------------------------------------------------------------------------
// Kernel 1: QKV projection. One wave = 16 pixels x ALL 192 outputs:
//  12 accumulators (q/k/v x 4 n-tiles) reuse each x A-fragment 12x, so x is
//  read exactly once from HBM. q pre-scaled by 1/sqrt(64).
// ---------------------------------------------------------------------------
__global__ __launch_bounds__(32) void spc_proj(
    const float* __restrict__ x,
    const float* __restrict__ Wq, const float* __restrict__ bq,
    const float* __restrict__ Wk, const float* __restrict__ bk,
    const float* __restrict__ Wv, const float* __restrict__ bv,
    __bf16* __restrict__ qo, __bf16* __restrict__ ko, __bf16* __restrict__ vo)
{
    int blk = blockIdx.x;
    int pt = blk & 255;
    int b  = blk >> 8;
    int lane = threadIdx.x & 31;
    int hl = lane >> 4, mn = lane & 15;
    int p0 = pt * 16;
    const float* xb = x + (size_t)b * CIN_ * HWQ;
    const float* Ws[3] = {Wq, Wk, Wv};
    const float* bs[3] = {bq, bk, bv};

    v8f acc[12];
#pragma unroll
    for (int t = 0; t < 12; ++t) {
        float bias = bs[t >> 2][(t & 3) * 16 + mn];
#pragma unroll
        for (int r = 0; r < 8; ++r) acc[t][r] = bias;
    }

    for (int kt = 0; kt < CIN_ / 32; ++kt) {
        int c0 = kt * 32;
        v16bf a;
#pragma unroll
        for (int e = 0; e < 16; ++e) {
            int kl = e + hl * 16;   // contiguous k order (matches B)
            a[e] = (__bf16)xb[(size_t)(c0 + kl) * HWQ + p0 + mn];
        }
#pragma unroll
        for (int t = 0; t < 12; ++t) {
            const float* wrow = Ws[t >> 2] + (size_t)((t & 3) * 16 + mn) * CIN_
                              + c0 + hl * 16;
            v16bf bf;
#pragma unroll
            for (int e = 0; e < 16; ++e) bf[e] = (__bf16)wrow[e];
            acc[t] = wmma_bf16(a, bf, acc[t]);
        }
    }

    __bf16* dsts[3] = {qo, ko, vo};
#pragma unroll
    for (int t = 0; t < 12; ++t) {
        __bf16* drow = dsts[t >> 2] + (size_t)b * HWQ * CMID;
        float scale = (t < 4) ? 0.125f : 1.0f;
#pragma unroll
        for (int r = 0; r < 8; ++r) {
            int m = r + hl * 8;
            drow[(size_t)(p0 + m) * CMID + (t & 3) * 16 + mn] =
                (__bf16)(acc[t][r] * scale);
        }
    }
}

// ---------------------------------------------------------------------------
// Kernel 2: streaming QK^T + online softmax stats. Per-lane private online
//  max/sum (NO cross-lane shuffles in the hot loop); single 16-lane merge at
//  the end. Diagonal score forced to 0 per reference.
// ---------------------------------------------------------------------------
__global__ __launch_bounds__(32) void spc_stats(
    const __bf16* __restrict__ q, const __bf16* __restrict__ k,
    float* __restrict__ Mo, float* __restrict__ Lo)
{
    int blk = blockIdx.x;
    int it = blk & 255;
    int b  = blk >> 8;
    int lane = threadIdx.x & 31;
    int hl = lane >> 4, mn = lane & 15;
    int i0 = it * 16;
    const __bf16* qb = q + (size_t)b * HWQ * CMID;
    const __bf16* kb = k + (size_t)b * HWQ * CMID;

    v16bf aq[2];
#pragma unroll
    for (int kt = 0; kt < 2; ++kt) {
        const __bf16* qrow = qb + (size_t)(i0 + mn) * CMID + kt * 32 + hl * 16;
        aq[kt] = cat8(*(const v8bf*)qrow, *(const v8bf*)(qrow + 8));
    }

    float mlane[8], llane[8];
#pragma unroll
    for (int r = 0; r < 8; ++r) { mlane[r] = -3.0e38f; llane[r] = 0.0f; }

    for (int jt = 0; jt < 256; ++jt) {
        int j0 = jt * 16;
        const __bf16* krow = kb + (size_t)(j0 + mn) * CMID + hl * 16;
        v16bf bk0 = *(const v16bf*)krow;
        v16bf bk1 = *(const v16bf*)(krow + 32);
        if (jt + 1 < 256)
            __builtin_prefetch(kb + (size_t)(j0 + 16 + mn) * CMID, 0, 0);
        v8f s = {};
        s = wmma_bf16(aq[0], bk0, s);
        s = wmma_bf16(aq[1], bk1, s);
        int gj = j0 + mn;
#pragma unroll
        for (int r = 0; r < 8; ++r) {
            int gi = i0 + r + hl * 8;
            float sv = (gi == gj) ? 0.0f : s[r];
            float mold = mlane[r];
            float mnew = fmaxf(mold, sv);
            llane[r] = llane[r] * __expf(mold - mnew) + __expf(sv - mnew);
            mlane[r] = mnew;
        }
    }

    // 16-lane merge: global max, rescale, sum.
#pragma unroll
    for (int r = 0; r < 8; ++r) {
        float m = mlane[r];
#pragma unroll
        for (int off = 8; off >= 1; off >>= 1)
            m = fmaxf(m, __shfl_xor(m, off, 16));
        float l = llane[r] * __expf(mlane[r] - m);
#pragma unroll
        for (int off = 8; off >= 1; off >>= 1)
            l += __shfl_xor(l, off, 16);
        if (mn == 0) {
            int gi = i0 + r + hl * 8;
            Mo[b * HWQ + gi] = m;
            Lo[b * HWQ + gi] = l;
        }
    }
}

// ---------------------------------------------------------------------------
// Kernel 3: output.
//  Phase 1: 49 window weights g = exp(s - M)/L into LDS (vectorized dots).
//  Phase 2: D[16x64] = A[16x3136] x B[3136x64] via 98x4 WMMAs. V rows are
//   fetched by TDM gather-mode (16 indexed rows x 128B per window) into
//   double-buffered LDS, DMA for chunk c+1 overlapped with WMMA on chunk c,
//   synchronized with s_wait_tensorcnt. Fallback: direct global loads.
// ---------------------------------------------------------------------------
__global__ __launch_bounds__(32) void spc_out(
    const __bf16* __restrict__ q, const __bf16* __restrict__ k,
    const __bf16* __restrict__ v,
    const float* __restrict__ Mi, const float* __restrict__ Li,
    const __bf16* __restrict__ Bt, const float* __restrict__ bout,
    float* __restrict__ out)
{
    __shared__ float  gS[16][49];
    __shared__ __bf16 vbuf[2][7][16][CMID];    // 28 KB double buffer

    int blk = blockIdx.x;
    int it = blk & 255;
    int b  = blk >> 8;
    int lane = threadIdx.x & 31;
    int hl = lane >> 4, mn = lane & 15;
    int i0 = it * 16;
    int h  = i0 >> 6;
    int w0 = i0 & 63;

    const __bf16* qb = q + (size_t)b * HWQ * CMID;
    const __bf16* kb = k + (size_t)b * HWQ * CMID;
    const __bf16* vb = v + (size_t)b * HWQ * CMID;

    // ---- Phase 1: windowed softmax weights --------------------------------
    for (int idx = lane; idx < 16 * 49; idx += 32) {
        int m  = idx / 49, wq = idx % 49;
        int dh = wq / 7 - 3, dw = wq % 7 - 3;
        int hh = min(max(h + dh, 0), 63);
        int ww = min(max(w0 + m + dw, 0), 63);
        int j  = hh * 64 + ww;
        int gi = i0 + m;
        float s = 0.0f;
        if (j != gi) {
            const v8bf* qr = (const v8bf*)(qb + (size_t)gi * CMID);
            const v8bf* kr = (const v8bf*)(kb + (size_t)j * CMID);
#pragma unroll
            for (int cc = 0; cc < 8; ++cc) {
                v8bf av = qr[cc], bv = kr[cc];
#pragma unroll
                for (int e = 0; e < 8; ++e) s += (float)av[e] * (float)bv[e];
            }
        }
        gS[m][wq] = __expf(s - Mi[b * HWQ + gi]) / Li[b * HWQ + gi];
    }
    __syncthreads();

    // ---- Phase 2: GEMM over K = 49*64 = 3136 ------------------------------
    v8f acc[4];
#pragma unroll
    for (int nt = 0; nt < 4; ++nt) {
        float bz = bout[nt * 16 + mn];
#pragma unroll
        for (int r = 0; r < 8; ++r) acc[nt][r] = bz;
    }

#if HAS_TDM
    unsigned vlds0 = (unsigned)(uintptr_t)&vbuf[0][0][0][0]; // LDS byte offset
    // Issue chunk 0 gathers (7 windows x 16 rows x 128B each).
    {
#pragma unroll
        for (int wl = 0; wl < 7; ++wl) {
            unsigned short idx[16];
            int dh = 0 - 3, dw = wl - 3;     // wq = wl for chunk 0
            int hh = min(max(h + dh, 0), 63);
#pragma unroll
            for (int m = 0; m < 16; ++m) {
                int ww = min(max(w0 + m + dw, 0), 63);
                idx[m] = (unsigned short)(hh * 64 + ww);
            }
            tdm_gather16(vlds0 + (unsigned)wl * 16 * CMID * 2, vb, idx);
        }
    }
#endif

    for (int ch = 0; ch < 7; ++ch) {
        int pbuf = ch & 1;
#if HAS_TDM
        if (ch + 1 < 7) {
#pragma unroll
            for (int wl = 0; wl < 7; ++wl) {
                unsigned short idx[16];
                int dh = (ch + 1) - 3, dw = wl - 3;
                int hh = min(max(h + dh, 0), 63);
#pragma unroll
                for (int m = 0; m < 16; ++m) {
                    int ww = min(max(w0 + m + dw, 0), 63);
                    idx[m] = (unsigned short)(hh * 64 + ww);
                }
                tdm_gather16(vlds0 + (unsigned)(((ch + 1) & 1) * 7 + wl)
                                         * 16 * CMID * 2, vb, idx);
            }
            __builtin_amdgcn_s_wait_tensorcnt(7);  // chunk ch done
        } else {
            __builtin_amdgcn_s_wait_tensorcnt(0);  // last chunk done
        }
#endif
#pragma unroll
        for (int wl = 0; wl < 7; ++wl) {
            int wq = ch * 7 + wl;
            float g = gS[mn][wq];
#pragma unroll
            for (int half = 0; half < 2; ++half) {
                int chalf = half * 32;
#if HAS_TDM
                const __bf16* vr = &vbuf[pbuf][wl][mn][chalf + hl * 16];
#else
                int dh = ch - 3, dw = wl - 3;
                int hh = min(max(h + dh, 0), 63);
                int ww = min(max(w0 + mn + dw, 0), 63);
                const __bf16* vr = vb + (size_t)(hh * 64 + ww) * CMID
                                 + chalf + hl * 16;
#endif
                v8bf lo = *(const v8bf*)vr;
                v8bf hi = *(const v8bf*)(vr + 8);
                v16bf a;
#pragma unroll
                for (int e = 0; e < 8; ++e) {
                    a[e]     = (__bf16)(g * (float)lo[e]);
                    a[e + 8] = (__bf16)(g * (float)hi[e]);
                }
                const __bf16* brow = Bt + (size_t)wq * 64 * 64 + chalf + hl * 16;
#pragma unroll
                for (int nt = 0; nt < 4; ++nt) {
                    v16bf bf = *(const v16bf*)(brow + (size_t)(nt * 16 + mn) * 64);
                    acc[nt] = wmma_bf16(a, bf, acc[nt]);
                }
            }
        }
    }

    // ---- Epilogue: ReLU + transposed store (b, o, h, w) -------------------
    float* ob = out + (size_t)b * CMID * HWQ;
#pragma unroll
    for (int nt = 0; nt < 4; ++nt) {
        int o = nt * 16 + mn;
#pragma unroll
        for (int r = 0; r < 8; ++r) {
            int m = r + hl * 8;
            ob[(size_t)o * HWQ + i0 + m] = fmaxf(acc[nt][r], 0.0f);
        }
    }
}

// ---------------------------------------------------------------------------
// Host-side launcher.
// ---------------------------------------------------------------------------
extern "C" void kernel_launch(void* const* d_in, const int* in_sizes, int n_in,
                              void* d_out, int out_size, void* d_ws, size_t ws_size,
                              hipStream_t stream) {
    (void)in_sizes; (void)n_in; (void)out_size; (void)ws_size;
    const float* x    = (const float*)d_in[0];
    const float* Wq   = (const float*)d_in[1];
    const float* bq   = (const float*)d_in[2];
    const float* Wk   = (const float*)d_in[3];
    const float* bk   = (const float*)d_in[4];
    const float* Wv   = (const float*)d_in[5];
    const float* bv   = (const float*)d_in[6];
    const float* Wout = (const float*)d_in[7];
    const float* bout = (const float*)d_in[8];
    float* out = (float*)d_out;

    char* ws = (char*)d_ws;
    const size_t QKV_BYTES = (size_t)NB * HWQ * CMID * sizeof(__bf16);  // 2 MB
    const size_t ML_BYTES  = (size_t)NB * HWQ * sizeof(float);          // 64 KB
    __bf16* qo = (__bf16*)(ws);
    __bf16* ko = (__bf16*)(ws + QKV_BYTES);
    __bf16* vo = (__bf16*)(ws + 2 * QKV_BYTES);
    float*  Mo = (float*)(ws + 3 * QKV_BYTES);
    float*  Lo = (float*)(ws + 3 * QKV_BYTES + ML_BYTES);
    __bf16* Bt = (__bf16*)(ws + 3 * QKV_BYTES + 2 * ML_BYTES);          // 392 KB

    spc_prep<<<(49 * 64 * 64 + 255) / 256, 256, 0, stream>>>(Wout, Bt);
    spc_proj<<<NB * 256, 32, 0, stream>>>(x, Wq, bq, Wk, bk, Wv, bv, qo, ko, vo);
    spc_stats<<<NB * 256, 32, 0, stream>>>(qo, ko, Mo, Lo);
    spc_out<<<NB * 256, 32, 0, stream>>>(qo, ko, vo, Mo, Lo, Bt, bout, out);
}
